// MyCustomHead_20959440404665
// MI455X (gfx1250) — matrix operand order repaired and checked by reading, not compiled, not run on hardware
//
#include <hip/hip_runtime.h>

// MI455X (gfx1250) fused routed-MLP head.
// Pipeline: f32->bf16 convert (+one-time weight transpose), counting-sort
// routing, 3 double-buffered bf16 WMMA GEMMs using GLOBAL_LOAD_ASYNC_TO_LDS
// in GVS mode (SGPR base + 32-bit lane offset), fused GEMV epilogue with
// deterministic partial reduction.

#define N_TOK   32768
#define D_IN    512
#define D_FF    1024
#define D_FFC   512
#define NTYPES  8

typedef __attribute__((ext_vector_type(16))) __bf16 v16bf;
typedef __attribute__((ext_vector_type(8)))  float  v8f;

union Frag16 {
    v16bf v;
    uint4 q[2];
};

// ---------------------------------------------------------- async LDS copy
// 16B per lane, memory -> LDS, ASYNCcnt-tracked. GVS mode: uniform SGPR base
// + 32-bit per-lane byte offset keeps VGPR pressure minimal. INST_OFFSET is
// added to BOTH the LDS and global address, which is exactly what the second
// 16B of a 32B row chunk needs.

__device__ __forceinline__ void async_cp16(unsigned lds_off, unsigned gvoff,
                                           const void* sbase) {
    asm volatile("global_load_async_to_lds_b128 %0, %1, %2"
                 :: "v"(lds_off), "v"(gvoff), "s"(sbase) : "memory");
}

__device__ __forceinline__ void async_cp16_o16(unsigned lds_off, unsigned gvoff,
                                               const void* sbase) {
    asm volatile("global_load_async_to_lds_b128 %0, %1, %2 offset:16"
                 :: "v"(lds_off), "v"(gvoff), "s"(sbase) : "memory");
}

__device__ __forceinline__ void wait_async() {
    asm volatile("s_wait_asynccnt 0" ::: "memory");
}

__device__ __forceinline__ unsigned lds_addr(const void* p) {
    return (unsigned)(uintptr_t)p;     // low 32 bits of generic addr = LDS offset
}

// ---------------------------------------------------------------- utilities

__global__ void zero_cnt_kernel(int* cnt) {
    if (threadIdx.x < NTYPES) cnt[threadIdx.x] = 0;
}

__global__ void cvt_f32_bf16(const float* __restrict__ s, __bf16* __restrict__ d, int n4) {
    int i = blockIdx.x * blockDim.x + threadIdx.x;
    if (i < n4) {
        float4 f = ((const float4*)s)[i];
        union { __bf16 e[4]; unsigned long long u; } p;
        p.e[0] = (__bf16)f.x; p.e[1] = (__bf16)f.y;
        p.e[2] = (__bf16)f.z; p.e[3] = (__bf16)f.w;
        ((unsigned long long*)d)[i] = p.u;
    }
}

// One-time weight convert + transpose: src f32 [K][N] -> dst bf16 [N][K].
__global__ void cvt_transpose_kernel(const float* __restrict__ src,
                                     __bf16* __restrict__ dst, int K, int N) {
    const size_t zs = (size_t)blockIdx.z * K * N;
    const float* s = src + zs;
    __bf16*      d = dst + zs;
    const int n  = blockIdx.x * 64 + (threadIdx.x & 63);
    const int k0 = blockIdx.y * 16 + (threadIdx.x >> 6) * 4;
    union { __bf16 e[4]; unsigned long long u; } p;
    #pragma unroll
    for (int j = 0; j < 4; ++j) p.e[j] = (__bf16)s[(size_t)(k0 + j) * N + n];
    *(unsigned long long*)&d[(size_t)n * K + k0] = p.u;
}

__global__ void histo_kernel(const int* __restrict__ types, int n, int* __restrict__ cnt) {
    int i = blockIdx.x * blockDim.x + threadIdx.x;
    if (i < n) atomicAdd(&cnt[types[i] & (NTYPES - 1)], 1);
}

__global__ void prefix_kernel(const int* __restrict__ cnt, int* __restrict__ off,
                              int* __restrict__ cur) {
    if (threadIdx.x == 0) {
        int s = 0;
        for (int t = 0; t < NTYPES; ++t) { off[t] = s; cur[t] = s; s += cnt[t]; }
        off[NTYPES] = s;
    }
}

__global__ void scatter_kernel(const int* __restrict__ types, int n,
                               int* __restrict__ cur, int* __restrict__ idx) {
    int i = blockIdx.x * blockDim.x + threadIdx.x;
    if (i < n) {
        int p = atomicAdd(&cur[types[i] & (NTYPES - 1)], 1);
        idx[p] = i;
    }
}

__global__ void reduce_out_kernel(const float* __restrict__ part,
                                  float* __restrict__ out, int n) {
    int i = blockIdx.x * blockDim.x + threadIdx.x;
    if (i < n) {
        float s = 0.f;
        #pragma unroll
        for (int j = 0; j < 8; ++j) s += part[(size_t)i * 8 + j];
        out[i] = s;
    }
}

// ------------------------------------------------------- GEMM 1: x @ W_pre
// C[N_TOK, D_FF] = relu(A[N_TOK, D_IN] @ B), B pre-transposed: BT[D_FF][D_IN].

__global__ __launch_bounds__(256) void gemm_pre(
    const __bf16* __restrict__ A,
    const __bf16* __restrict__ BT,
    __bf16* __restrict__ C)
{
    __shared__ __align__(16) __bf16 As[2][128][40];  // 80B rows: aligned, conflict-spread
    __shared__ __align__(16) __bf16 Bs[2][128][40];

    const int tid  = threadIdx.x;
    const int lane = tid & 31;
    const int wid  = tid >> 5;
    const int wm   = wid & 3;        // 4 waves along M (32 rows each)
    const int wn   = wid >> 2;       // 2 waves along N (64 cols each)
    const int lm   = lane & 15;
    const int lk   = lane >> 4;

    const int m0 = blockIdx.y * 128;
    const int n0 = blockIdx.x * 128;

    const v8f vz = {0.f,0.f,0.f,0.f,0.f,0.f,0.f,0.f};
    v8f acc[2][4];
    for (int i = 0; i < 2; ++i)
        for (int j = 0; j < 4; ++j) acc[i][j] = vz;

    const int row = tid >> 1;            // tile row (A: m, B: n), K-contiguous
    const int col = (tid & 1) * 16;
    const unsigned voffA = ((unsigned)(m0 + row) * D_IN + col) * 2u;  // bytes
    const unsigned voffB = ((unsigned)(n0 + row) * D_IN + col) * 2u;
    const unsigned ldsA[2] = { lds_addr(&As[0][row][col]), lds_addr(&As[1][row][col]) };
    const unsigned ldsB[2] = { lds_addr(&Bs[0][row][col]), lds_addr(&Bs[1][row][col]) };

    auto load_tile = [&](int b, int kc) {
        const unsigned ka = voffA + (unsigned)kc * 2u;
        const unsigned kb = voffB + (unsigned)kc * 2u;
        async_cp16    (ldsA[b], ka, A);
        async_cp16_o16(ldsA[b], ka, A);
        async_cp16    (ldsB[b], kb, BT);
        async_cp16_o16(ldsB[b], kb, BT);
    };

    const int NIT = D_IN / 32;
    load_tile(0, 0);
    for (int it = 0; it < NIT; ++it) {
        wait_async();                    // this wave's fills of buffer (it&1) done
        __syncthreads();                 // everyone's fills visible; prev reads done
        if (it + 1 < NIT) load_tile((it + 1) & 1, (it + 1) * 32);  // overlap w/ WMMA

        const int b = it & 1;
        Frag16 a[2], bb[4];
        #pragma unroll
        for (int i = 0; i < 2; ++i) {
            const int r = wm * 32 + i * 16 + lm;
            a[i].q[0] = *(const uint4*)&As[b][r][lk * 8];
            a[i].q[1] = *(const uint4*)&As[b][r][16 + lk * 8];
        }
        #pragma unroll
        for (int j = 0; j < 4; ++j) {
            const int c = wn * 64 + j * 16 + lm;
            bb[j].q[0] = *(const uint4*)&Bs[b][c][lk * 8];
            bb[j].q[1] = *(const uint4*)&Bs[b][c][16 + lk * 8];
        }
        #pragma unroll
        for (int i = 0; i < 2; ++i)
            #pragma unroll
            for (int j = 0; j < 4; ++j)
                acc[i][j] = __builtin_amdgcn_wmma_f32_16x16x32_bf16(
                    false, a[i].v, false, bb[j].v, (short)0, acc[i][j], false, false);
    }

    #pragma unroll
    for (int i = 0; i < 2; ++i)
        #pragma unroll
        for (int j = 0; j < 4; ++j) {
            const int n = n0 + wn * 64 + j * 16 + lm;
            #pragma unroll
            for (int v = 0; v < 8; ++v) {
                const int m = m0 + wm * 32 + i * 16 + v + 8 * lk;
                float x = acc[i][j][v];
                C[(size_t)m * D_FF + n] = (__bf16)(x > 0.f ? x : 0.f);
            }
        }
}

// --------------------------------------------- GEMM 2: routed per-type MLP
// For sorted positions p in this 128-tile: tok = idx[p], t = type(tok)
// Xr[tok] = bf16( Xres[tok] + relu(Aff[tok] @ W_types[t]) ), WtT pre-transposed.

__global__ __launch_bounds__(256) void gemm_typed(
    const __bf16* __restrict__ Aff,   // [N_TOK, D_FF]
    const __bf16* __restrict__ WtT,   // [NTYPES, D_IN, D_FF]  (transposed)
    const float*  __restrict__ Xres,  // [N_TOK, D_IN]
    const int*    __restrict__ idx,   // [N_TOK]
    const int*    __restrict__ off,   // [NTYPES+1]
    __bf16* __restrict__ Xr)          // [N_TOK, D_IN]
{
    __shared__ __align__(16) __bf16 As[2][128][40];
    __shared__ __align__(16) __bf16 Bs[2][128][40];
    __shared__ int idxs[128];

    const int tid  = threadIdx.x;
    const int lane = tid & 31;
    const int wid  = tid >> 5;
    const int wm   = wid & 3;
    const int wn   = wid >> 2;
    const int lm   = lane & 15;
    const int lk   = lane >> 4;

    const int p0 = blockIdx.y * 128;
    const int n0 = blockIdx.x * 128;

    if (tid < 128) idxs[tid] = idx[p0 + tid];
    __syncthreads();

    const int row = tid >> 1;
    const int col = (tid & 1) * 16;
    const int mytok = idxs[row];                       // gathered A row
    const unsigned voffA = ((unsigned)mytok * D_FF + col) * 2u;         // <= 67MB, fits i32
    const unsigned voffB = ((unsigned)(n0 + row) * D_FF + col) * 2u;
    const unsigned ldsA[2] = { lds_addr(&As[0][row][col]), lds_addr(&As[1][row][col]) };
    const unsigned ldsB[2] = { lds_addr(&Bs[0][row][col]), lds_addr(&Bs[1][row][col]) };

    const v8f vz = {0.f,0.f,0.f,0.f,0.f,0.f,0.f,0.f};

    for (int t = 0; t < NTYPES; ++t) {                 // block-uniform loop
        int lo = off[t];     if (lo < p0)       lo = p0;
        int hi = off[t + 1]; if (hi > p0 + 128) hi = p0 + 128;
        if (lo >= hi) continue;                        // type absent from tile

        const __bf16* Bt = WtT + (size_t)t * D_IN * D_FF;   // uniform SGPR base

        v8f acc[2][4];
        for (int i = 0; i < 2; ++i)
            for (int j = 0; j < 4; ++j) acc[i][j] = vz;

        auto load_tile = [&](int b, int kc) {
            const unsigned ka = voffA + (unsigned)kc * 2u;
            const unsigned kb = voffB + (unsigned)kc * 2u;
            async_cp16    (ldsA[b], ka, Aff);
            async_cp16_o16(ldsA[b], ka, Aff);
            async_cp16    (ldsB[b], kb, Bt);
            async_cp16_o16(ldsB[b], kb, Bt);
        };

        const int NIT = D_FF / 32;
        __syncthreads();                               // buffers free from prev type
        load_tile(0, 0);
        for (int it = 0; it < NIT; ++it) {
            wait_async();
            __syncthreads();
            if (it + 1 < NIT) load_tile((it + 1) & 1, (it + 1) * 32);

            const int b = it & 1;
            Frag16 a[2], bb[4];
            #pragma unroll
            for (int i = 0; i < 2; ++i) {
                const int r = wm * 32 + i * 16 + lm;
                a[i].q[0] = *(const uint4*)&As[b][r][lk * 8];
                a[i].q[1] = *(const uint4*)&As[b][r][16 + lk * 8];
            }
            #pragma unroll
            for (int j = 0; j < 4; ++j) {
                const int c = wn * 64 + j * 16 + lm;
                bb[j].q[0] = *(const uint4*)&Bs[b][c][lk * 8];
                bb[j].q[1] = *(const uint4*)&Bs[b][c][16 + lk * 8];
            }
            #pragma unroll
            for (int i = 0; i < 2; ++i)
                #pragma unroll
                for (int j = 0; j < 4; ++j)
                    acc[i][j] = __builtin_amdgcn_wmma_f32_16x16x32_bf16(
                        false, a[i].v, false, bb[j].v, (short)0, acc[i][j], false, false);
        }

        // relu + residual, store only rows owned by this type
        #pragma unroll
        for (int i = 0; i < 2; ++i)
            #pragma unroll
            for (int j = 0; j < 4; ++j) {
                const int n = n0 + wn * 64 + j * 16 + lm;
                #pragma unroll
                for (int v = 0; v < 8; ++v) {
                    const int pl = wm * 32 + i * 16 + v + 8 * lk;
                    const int p  = p0 + pl;
                    if (p >= lo && p < hi) {
                        const int tok = idxs[pl];
                        float x = acc[i][j][v];
                        x = (x > 0.f ? x : 0.f) + Xres[(size_t)tok * D_IN + n];
                        Xr[(size_t)tok * D_IN + n] = (__bf16)x;
                    }
                }
            }
    }
}

// ------------------------------------- GEMM 3 + fused GEMV: relu(.@Wc1)@Wc2
// part[m][slot] = sum over this wave's 64 N-cols of relu(h[m][n]) * Wc2[n]

__global__ __launch_bounds__(256) void gemm_head(
    const __bf16* __restrict__ Xr,    // [N_TOK, D_IN]
    const __bf16* __restrict__ Wc1T,  // [D_FFC, D_IN] (transposed)
    const float*  __restrict__ Wc2,   // [D_FFC]
    float* __restrict__ part)         // [N_TOK, 8]
{
    __shared__ __align__(16) __bf16 As[2][128][40];
    __shared__ __align__(16) __bf16 Bs[2][128][40];

    const int tid  = threadIdx.x;
    const int lane = tid & 31;
    const int wid  = tid >> 5;
    const int wm   = wid & 3;
    const int wn   = wid >> 2;
    const int lm   = lane & 15;
    const int lk   = lane >> 4;

    const int m0 = blockIdx.y * 128;
    const int n0 = blockIdx.x * 128;

    const v8f vz = {0.f,0.f,0.f,0.f,0.f,0.f,0.f,0.f};
    v8f acc[2][4];
    for (int i = 0; i < 2; ++i)
        for (int j = 0; j < 4; ++j) acc[i][j] = vz;

    const int row = tid >> 1;
    const int col = (tid & 1) * 16;
    const unsigned voffA = ((unsigned)(m0 + row) * D_IN + col) * 2u;
    const unsigned voffB = ((unsigned)(n0 + row) * D_IN + col) * 2u;
    const unsigned ldsA[2] = { lds_addr(&As[0][row][col]), lds_addr(&As[1][row][col]) };
    const unsigned ldsB[2] = { lds_addr(&Bs[0][row][col]), lds_addr(&Bs[1][row][col]) };

    auto load_tile = [&](int b, int kc) {
        const unsigned ka = voffA + (unsigned)kc * 2u;
        const unsigned kb = voffB + (unsigned)kc * 2u;
        async_cp16    (ldsA[b], ka, Xr);
        async_cp16_o16(ldsA[b], ka, Xr);
        async_cp16    (ldsB[b], kb, Wc1T);
        async_cp16_o16(ldsB[b], kb, Wc1T);
    };

    const int NIT = D_IN / 32;
    load_tile(0, 0);
    for (int it = 0; it < NIT; ++it) {
        wait_async();
        __syncthreads();
        if (it + 1 < NIT) load_tile((it + 1) & 1, (it + 1) * 32);

        const int b = it & 1;
        Frag16 a[2], bb[4];
        #pragma unroll
        for (int i = 0; i < 2; ++i) {
            const int r = wm * 32 + i * 16 + lm;
            a[i].q[0] = *(const uint4*)&As[b][r][lk * 8];
            a[i].q[1] = *(const uint4*)&As[b][r][16 + lk * 8];
        }
        #pragma unroll
        for (int j = 0; j < 4; ++j) {
            const int c = wn * 64 + j * 16 + lm;
            bb[j].q[0] = *(const uint4*)&Bs[b][c][lk * 8];
            bb[j].q[1] = *(const uint4*)&Bs[b][c][16 + lk * 8];
        }
        #pragma unroll
        for (int i = 0; i < 2; ++i)
            #pragma unroll
            for (int j = 0; j < 4; ++j)
                acc[i][j] = __builtin_amdgcn_wmma_f32_16x16x32_bf16(
                    false, a[i].v, false, bb[j].v, (short)0, acc[i][j], false, false);
    }

    // fused GEMV epilogue: relu, scale by Wc2, reduce over the 16 N-lanes
    float w2[4];
    #pragma unroll
    for (int j = 0; j < 4; ++j) w2[j] = Wc2[n0 + wn * 64 + j * 16 + lm];

    const int slot = blockIdx.x * 2 + wn;        // 4 N-blocks x 2 N-waves = 8 slots
    #pragma unroll
    for (int i = 0; i < 2; ++i) {
        #pragma unroll
        for (int v = 0; v < 8; ++v) {
            float s = 0.f;
            #pragma unroll
            for (int j = 0; j < 4; ++j) {
                float x = acc[i][j][v];
                s += (x > 0.f ? x : 0.f) * w2[j];
            }
            s += __shfl_xor(s, 1, 16);
            s += __shfl_xor(s, 2, 16);
            s += __shfl_xor(s, 4, 16);
            s += __shfl_xor(s, 8, 16);
            if (lm == 0) {
                const int m = m0 + wm * 32 + i * 16 + v + 8 * lk;
                part[(size_t)m * 8 + slot] = s;  // unique (m, slot): deterministic
            }
        }
    }
}

// ---------------------------------------------------------------- launcher

extern "C" void kernel_launch(void* const* d_in, const int* in_sizes, int n_in,
                              void* d_out, int out_size, void* d_ws, size_t ws_size,
                              hipStream_t stream)
{
    (void)in_sizes; (void)n_in; (void)out_size; (void)ws_size;

    const float* x     = (const float*)d_in[0];
    const int*   types = (const int*)  d_in[1];
    const float* Wpre  = (const float*)d_in[2];
    const float* Wtyp  = (const float*)d_in[3];
    const float* Wc1   = (const float*)d_in[4];
    const float* Wc2   = (const float*)d_in[5];
    float* out = (float*)d_out;

    char* w = (char*)d_ws;
    auto take = [&](size_t bytes) -> char* {
        char* p = w;
        w += (bytes + 255) & ~(size_t)255;
        return p;
    };
    __bf16* xbf   = (__bf16*)take((size_t)N_TOK * D_IN * 2);
    __bf16* wpreT = (__bf16*)take((size_t)D_IN * D_FF * 2);          // [D_FF][D_IN]
    __bf16* wtypT = (__bf16*)take((size_t)NTYPES * D_FF * D_IN * 2); // [t][D_IN][D_FF]
    __bf16* wc1T  = (__bf16*)take((size_t)D_IN * D_FFC * 2);         // [D_FFC][D_IN]
    __bf16* xff   = (__bf16*)take((size_t)N_TOK * D_FF * 2);
    __bf16* xr    = (__bf16*)take((size_t)N_TOK * D_IN * 2);
    float*  part  = (float*) take((size_t)N_TOK * 8 * 4);
    int*    idx   = (int*)   take((size_t)N_TOK * 4);
    int*    cnt   = (int*)   take(64);
    int*    off   = (int*)   take(64);
    int*    cur   = (int*)   take(64);

    zero_cnt_kernel<<<1, 32, 0, stream>>>(cnt);

    int n4 = N_TOK * D_IN / 4;
    cvt_f32_bf16<<<(n4 + 255) / 256, 256, 0, stream>>>(x, xbf, n4);

    // one-time weight convert + transpose (K rows, N cols -> [N][K] bf16)
    cvt_transpose_kernel<<<dim3(D_FF / 64, D_IN / 16, 1),      256, 0, stream>>>(Wpre, wpreT, D_IN, D_FF);
    cvt_transpose_kernel<<<dim3(D_IN / 64, D_FF / 16, NTYPES), 256, 0, stream>>>(Wtyp, wtypT, D_FF, D_IN);
    cvt_transpose_kernel<<<dim3(D_FFC / 64, D_IN / 16, 1),     256, 0, stream>>>(Wc1, wc1T, D_IN, D_FFC);

    histo_kernel  <<<N_TOK / 256, 256, 0, stream>>>(types, N_TOK, cnt);
    prefix_kernel <<<1, 32, 0, stream>>>(cnt, off, cur);
    scatter_kernel<<<N_TOK / 256, 256, 0, stream>>>(types, N_TOK, cur, idx);

    gemm_pre  <<<dim3(D_FF  / 128, N_TOK / 128), 256, 0, stream>>>(xbf, wpreT, xff);
    gemm_typed<<<dim3(D_IN  / 128, N_TOK / 128), 256, 0, stream>>>(xff, wtypT, x, idx, off, xr);
    gemm_head <<<dim3(D_FFC / 128, N_TOK / 128), 256, 0, stream>>>(xr, wc1T, Wc2, part);

    reduce_out_kernel<<<(N_TOK + 255) / 256, 256, 0, stream>>>(part, out, N_TOK);
}
